// GNNEncoder_85306640433195
// MI455X (gfx1250) — matrix operand order repaired and checked by reading, not compiled, not run on hardware
//
#include <hip/hip_runtime.h>
#include <hip/hip_bf16.h>

#define NS 100000
#define NI 20000
#define DD 128
#define NE1 600000
#define NE2 500000
#define NL 2

typedef __bf16 bf16_t;
typedef bf16_t bvec16 __attribute__((ext_vector_type(16)));
typedef bf16_t bvec8  __attribute__((ext_vector_type(8)));
typedef float  fvec8  __attribute__((ext_vector_type(8)));
typedef int    v4i    __attribute__((ext_vector_type(4)));

// LDS row stride for 16x128 bf16 A tiles: 128 + 8 pad elements = 272 B/row.
// Each row shifts by 4 banks -> 16 rows cover all 64 banks, conflict-free
// ds_load_b128 fragment reads.
#define LDS_STRIDE 136

// ---------------- async global->LDS copy (CDNA5) ----------------------------
#if defined(__has_builtin)
#if __has_builtin(__builtin_amdgcn_global_load_async_to_lds_b128)
#define HAVE_ASYNC_LDS 1
#endif
#endif

// builtin signature (from clang diagnostic): (int4 AS1*, int4 AS3*, Ii, Ii)
typedef __attribute__((address_space(1))) v4i* as1_v4i_p;
typedef __attribute__((address_space(3))) v4i* as3_v4i_p;

__device__ __forceinline__ void copy16_to_lds(const void* g, void* l) {
#ifdef HAVE_ASYNC_LDS
    // per-lane 16B global->LDS DMA, tracked by ASYNCcnt
    __builtin_amdgcn_global_load_async_to_lds_b128(
        (as1_v4i_p)(uintptr_t)g, (as3_v4i_p)(unsigned)(uintptr_t)l, 0, 0);
#else
    *(bvec8*)l = *(const bvec8*)g;  // fallback: bounce through VGPRs
#endif
}

__device__ __forceinline__ void async_wait_all() {
#ifdef HAVE_ASYNC_LDS
#if __has_builtin(__builtin_amdgcn_s_wait_asynccnt)
    __builtin_amdgcn_s_wait_asynccnt(0);
#else
    asm volatile("s_wait_asynccnt 0" ::: "memory");
#endif
#endif
}

// Stage a contiguous 16x128 bf16 tile (4 KB) into padded LDS; one 16B chunk
// per thread (256 threads).
__device__ __forceinline__ void stage_a_tile(const bf16_t* __restrict__ g,
                                             bf16_t* lds, int tid) {
    const int row = tid >> 4;        // 0..15
    const int seg = (tid & 15) * 8;  // element offset, 8 bf16 = 16 B
    copy16_to_lds(g + row * DD + seg, lds + row * LDS_STRIDE + seg);
}

// ---------------- WMMA fragment loaders (gfx1250 wave32 layouts) ------------
// A fragment from padded LDS tile.
// lane<16 : row=lane,    K = k0+{0..15}
// lane>=16: row=lane-16, K = k0+{16..31}
__device__ __forceinline__ bvec16 lds_a_frag(const bf16_t* lds, int k0, int lane) {
    const int row = lane & 15;
    const int kh  = (lane >> 4) * 16;
    const bf16_t* p = lds + row * LDS_STRIDE + k0 + kh;
    union { bvec16 v; bvec8 h[2]; } u;
    u.h[0] = *(const bvec8*)(p);
    u.h[1] = *(const bvec8*)(p + 8);
    return u.v;
}

// B: 32x16 bf16 tile of W^T == 16 rows of W. Lane n holds column n (a row of
// W), 16 contiguous K values starting at k0 (+16 for lanes 16..31).
__device__ __forceinline__ bvec16 load_b_frag(const bf16_t* __restrict__ W,
                                              int k0, int lane) {
    const int col = lane & 15;
    const int kh  = (lane >> 4) * 16;
    return *(const bvec16*)(W + (size_t)col * DD + k0 + kh);
}

#define WMMA_BF16(acc, a, b) \
    __builtin_amdgcn_wmma_f32_16x16x32_bf16(false, (a), false, (b), (short)0, (acc), false, false)

// ---------------- GEMM kernels ---------------------------------------------
// grid.x = M/16 row tiles, 256 threads = 8 waves = 8 column tiles (N = 128).

// h = X @ W^T   (no bias), f32 out
__global__ __launch_bounds__(256)
void gemm_plain(const bf16_t* __restrict__ X, const bf16_t* __restrict__ W,
                float* __restrict__ out) {
    __shared__ bf16_t sA[16 * LDS_STRIDE];
    const int wave = threadIdx.x >> 5;
    const int lane = threadIdx.x & 31;
    const int rowbase = blockIdx.x * 16;
    const int colbase = wave * 16;

    stage_a_tile(X + (size_t)rowbase * DD, sA, threadIdx.x);
    async_wait_all();
    __syncthreads();

    const bf16_t* B = W + (size_t)colbase * DD;
    fvec8 acc = {};
#pragma unroll
    for (int k0 = 0; k0 < DD; k0 += 32) {
        acc = WMMA_BF16(acc, lds_a_frag(sA, k0, lane), load_b_frag(B, k0, lane));
    }
    const int n  = lane & 15;
    const int mo = (lane >> 4) * 8;
#pragma unroll
    for (int r = 0; r < 8; ++r)
        out[(size_t)(rowbase + mo + r) * DD + colbase + n] = acc[r];
}

// item = elu(agg @ Wl^T + xi @ Wr^T + bl)
__global__ __launch_bounds__(256)
void gemm_sage_item(const bf16_t* __restrict__ Agg, const bf16_t* __restrict__ Xd,
                    const bf16_t* __restrict__ Wl, const bf16_t* __restrict__ Wr,
                    const float* __restrict__ bl, float* __restrict__ out) {
    __shared__ bf16_t sA1[16 * LDS_STRIDE];
    __shared__ bf16_t sA2[16 * LDS_STRIDE];
    const int wave = threadIdx.x >> 5;
    const int lane = threadIdx.x & 31;
    const int rowbase = blockIdx.x * 16;
    const int colbase = wave * 16;

    stage_a_tile(Agg + (size_t)rowbase * DD, sA1, threadIdx.x);
    stage_a_tile(Xd  + (size_t)rowbase * DD, sA2, threadIdx.x);
    async_wait_all();
    __syncthreads();

    const bf16_t* B1 = Wl + (size_t)colbase * DD;
    const bf16_t* B2 = Wr + (size_t)colbase * DD;
    fvec8 acc = {};
#pragma unroll
    for (int k0 = 0; k0 < DD; k0 += 32) {
        acc = WMMA_BF16(acc, lds_a_frag(sA1, k0, lane), load_b_frag(B1, k0, lane));
        acc = WMMA_BF16(acc, lds_a_frag(sA2, k0, lane), load_b_frag(B2, k0, lane));
    }
    const int n  = lane & 15;
    const int mo = (lane >> 4) * 8;
    const float bias = bl[colbase + n];
#pragma unroll
    for (int r = 0; r < 8; ++r) {
        float v = acc[r] + bias;
        v = v > 0.f ? v : (__expf(v) - 1.f);           // ELU (alpha=1)
        out[(size_t)(rowbase + mo + r) * DD + colbase + n] = v;
    }
}

// stu = 0.5*( agg @ Wl^T + xs @ Wr^T + bl  +  simple_acc + b_p )
__global__ __launch_bounds__(256)
void gemm_sage_stu(const bf16_t* __restrict__ Agg, const bf16_t* __restrict__ Xd,
                   const bf16_t* __restrict__ Wl, const bf16_t* __restrict__ Wr,
                   const float* __restrict__ bl, const float* __restrict__ simple,
                   const float* __restrict__ bp, float* __restrict__ out) {
    __shared__ bf16_t sA1[16 * LDS_STRIDE];
    __shared__ bf16_t sA2[16 * LDS_STRIDE];
    const int wave = threadIdx.x >> 5;
    const int lane = threadIdx.x & 31;
    const int rowbase = blockIdx.x * 16;
    const int colbase = wave * 16;

    stage_a_tile(Agg + (size_t)rowbase * DD, sA1, threadIdx.x);
    stage_a_tile(Xd  + (size_t)rowbase * DD, sA2, threadIdx.x);
    async_wait_all();
    __syncthreads();

    const bf16_t* B1 = Wl + (size_t)colbase * DD;
    const bf16_t* B2 = Wr + (size_t)colbase * DD;
    fvec8 acc = {};
#pragma unroll
    for (int k0 = 0; k0 < DD; k0 += 32) {
        acc = WMMA_BF16(acc, lds_a_frag(sA1, k0, lane), load_b_frag(B1, k0, lane));
        acc = WMMA_BF16(acc, lds_a_frag(sA2, k0, lane), load_b_frag(B2, k0, lane));
    }
    const int n  = lane & 15;
    const int mo = (lane >> 4) * 8;
    const float bsum = bl[colbase + n] + bp[colbase + n];
#pragma unroll
    for (int r = 0; r < 8; ++r) {
        const size_t idx = (size_t)(rowbase + mo + r) * DD + colbase + n;
        out[idx] = 0.5f * (acc[r] + bsum + simple[idx]);
    }
}

// ---------------- scatter / gather -----------------------------------------
// one wave per edge; 128 floats per row => 4 coalesced atomics per lane
__global__ __launch_bounds__(256)
void scatter_mean_acc(const float* __restrict__ x, const int* __restrict__ src,
                      const int* __restrict__ dst, float* __restrict__ agg,
                      float* __restrict__ cnt, int E) {
    const int e = (int)((blockIdx.x * 256u + threadIdx.x) >> 5);
    const int lane = threadIdx.x & 31;
    if (e >= E) return;
    const int s = src[e], d = dst[e];
    const float* xr = x + (size_t)s * DD;
    float* ar = agg + (size_t)d * DD;
#pragma unroll
    for (int i = 0; i < 4; ++i)
        atomicAdd(ar + lane + i * 32, xr[lane + i * 32]);
    if (lane == 0) atomicAdd(cnt + d, 1.0f);
}

__global__ __launch_bounds__(256)
void scatter_add(const float* __restrict__ x, const int* __restrict__ src,
                 const int* __restrict__ dst, float* __restrict__ acc, int E) {
    const int e = (int)((blockIdx.x * 256u + threadIdx.x) >> 5);
    const int lane = threadIdx.x & 31;
    if (e >= E) return;
    const int s = src[e], d = dst[e];
    const float* xr = x + (size_t)s * DD;
    float* ar = acc + (size_t)d * DD;
#pragma unroll
    for (int i = 0; i < 4; ++i)
        atomicAdd(ar + lane + i * 32, xr[lane + i * 32]);
}

// ---------------- elementwise helpers --------------------------------------
__global__ void f32_to_bf16(const float* __restrict__ in, bf16_t* __restrict__ out, int n) {
    const int i = blockIdx.x * blockDim.x + threadIdx.x;
    if (i < n) out[i] = (bf16_t)in[i];
}

// agg / clip(cnt,1)  ->  bf16
__global__ void norm_to_bf16(const float* __restrict__ agg, const float* __restrict__ cnt,
                             bf16_t* __restrict__ out, int n) {
    const int i = blockIdx.x * blockDim.x + threadIdx.x;
    if (i >= n) return;
    float c = cnt[i >> 7];
    c = c < 1.f ? 1.f : c;
    out[i] = (bf16_t)(agg[i] / c);
}

__global__ __launch_bounds__(256)
void bn_stats(const float* __restrict__ x, float* __restrict__ sum,
              float* __restrict__ sumsq, int nrows) {
    const int col  = threadIdx.x & 127;
    const int rofs = threadIdx.x >> 7;  // 0..1
    const int rowbase = blockIdx.x * 128;
    float s = 0.f, q = 0.f;
    for (int r = rofs; r < 128 && rowbase + r < nrows; r += 2) {
        const float v = x[(size_t)(rowbase + r) * DD + col];
        s += v; q += v * v;
    }
    atomicAdd(sum + col, s);
    atomicAdd(sumsq + col, q);
}

__global__ void bn_apply(const float* __restrict__ x, const float* __restrict__ sum,
                         const float* __restrict__ sumsq, const float* __restrict__ g,
                         const float* __restrict__ b, float* __restrict__ out, int nrows) {
    const int i = blockIdx.x * blockDim.x + threadIdx.x;
    if (i >= nrows * DD) return;
    const int col = i & 127;
    const float inv = 1.0f / (float)nrows;
    const float m = sum[col] * inv;
    const float v = sumsq[col] * inv - m * m;  // biased variance
    out[i] = (x[i] - m) * rsqrtf(v + 1e-5f) * g[col] + b[col];
}

// ---------------- driver ----------------------------------------------------
extern "C" void kernel_launch(void* const* d_in, const int* in_sizes, int n_in,
                              void* d_out, int out_size, void* d_ws, size_t ws_size,
                              hipStream_t stream) {
    const float* x_student = (const float*)d_in[0];
    const float* x_item    = (const float*)d_in[1];
    const float* Wl_ri = (const float*)d_in[2];
    const float* bl_ri = (const float*)d_in[3];
    const float* Wr_ri = (const float*)d_in[4];
    const float* Wl_rs = (const float*)d_in[5];
    const float* bl_rs = (const float*)d_in[6];
    const float* Wr_rs = (const float*)d_in[7];
    const float* W_p   = (const float*)d_in[8];
    const float* b_p   = (const float*)d_in[9];
    const float* bn_g  = (const float*)d_in[10];
    const float* bn_b  = (const float*)d_in[11];
    const int* resp_src = (const int*)d_in[12];
    const int* resp_dst = (const int*)d_in[13];
    const int* rev_src  = (const int*)d_in[14];
    const int* rev_dst  = (const int*)d_in[15];
    const int* prec_src = (const int*)d_in[16];
    const int* prec_dst = (const int*)d_in[17];

    // ---- workspace carve-up ----
    char* w = (char*)d_ws;
    auto carve = [&](size_t bytes) {
        void* p = (void*)w;
        w += (bytes + 255) & ~(size_t)255;
        return p;
    };
    float* xs_buf     = (float*)carve((size_t)NS * DD * 4);  // layer-1 student out
    float* xi_buf     = (float*)carve((size_t)NI * DD * 4);  // layer-1 item out
    float* agg_s      = (float*)carve((size_t)NS * DD * 4);  // also reused as h_p
    float* agg_i      = (float*)carve((size_t)NI * DD * 4);  // also reused as elu(item)
    float* simple_acc = (float*)carve((size_t)NS * DD * 4);
    float* cnt_s      = (float*)carve((size_t)NS * 4);
    float* cnt_i      = (float*)carve((size_t)NI * 4);
    float* bnsum      = (float*)carve((size_t)DD * 4);
    float* bnsq       = (float*)carve((size_t)DD * 4);
    bf16_t* xs_bf   = (bf16_t*)carve((size_t)NS * DD * 2);
    bf16_t* xi_bf   = (bf16_t*)carve((size_t)NI * DD * 2);
    bf16_t* aggs_bf = (bf16_t*)carve((size_t)NS * DD * 2);
    bf16_t* aggi_bf = (bf16_t*)carve((size_t)NI * DD * 2);
    bf16_t* Wlri_bf = (bf16_t*)carve((size_t)DD * DD * 2);
    bf16_t* Wrri_bf = (bf16_t*)carve((size_t)DD * DD * 2);
    bf16_t* Wlrs_bf = (bf16_t*)carve((size_t)DD * DD * 2);
    bf16_t* Wrrs_bf = (bf16_t*)carve((size_t)DD * DD * 2);
    bf16_t* Wp_bf   = (bf16_t*)carve((size_t)DD * DD * 2);

    const int EL = 256;
    const int gNS = (NS * DD + EL - 1) / EL;
    const int gNI = (NI * DD + EL - 1) / EL;
    const int gW  = (DD * DD + EL - 1) / EL;

    for (int l = 0; l < NL; ++l) {
        const float* xs_in = (l == 0) ? x_student : xs_buf;
        const float* xi_in = (l == 0) ? x_item    : xi_buf;
        float* xi_out = (l == NL - 1) ? (float*)d_out                  : xi_buf;
        float* xs_out = (l == NL - 1) ? (float*)d_out + (size_t)NI * DD : xs_buf;

        const size_t wofs = (size_t)l * DD * DD;
        const size_t bofs = (size_t)l * DD;

        // zero accumulators
        (void)hipMemsetAsync(agg_i,      0, (size_t)NI * DD * 4, stream);
        (void)hipMemsetAsync(cnt_i,      0, (size_t)NI * 4,      stream);
        (void)hipMemsetAsync(agg_s,      0, (size_t)NS * DD * 4, stream);
        (void)hipMemsetAsync(cnt_s,      0, (size_t)NS * 4,      stream);
        (void)hipMemsetAsync(simple_acc, 0, (size_t)NS * DD * 4, stream);
        (void)hipMemsetAsync(bnsum,      0, (size_t)DD * 4,      stream);
        (void)hipMemsetAsync(bnsq,       0, (size_t)DD * 4,      stream);

        // bf16 copies of features + this layer's weights
        f32_to_bf16<<<gNS, EL, 0, stream>>>(xs_in, xs_bf, NS * DD);
        f32_to_bf16<<<gNI, EL, 0, stream>>>(xi_in, xi_bf, NI * DD);
        f32_to_bf16<<<gW, EL, 0, stream>>>(Wl_ri + wofs, Wlri_bf, DD * DD);
        f32_to_bf16<<<gW, EL, 0, stream>>>(Wr_ri + wofs, Wrri_bf, DD * DD);
        f32_to_bf16<<<gW, EL, 0, stream>>>(Wl_rs + wofs, Wlrs_bf, DD * DD);
        f32_to_bf16<<<gW, EL, 0, stream>>>(Wr_rs + wofs, Wrrs_bf, DD * DD);
        f32_to_bf16<<<gW, EL, 0, stream>>>(W_p   + wofs, Wp_bf,   DD * DD);

        // segment sums for both SAGE directions
        scatter_mean_acc<<<(NE1 + 7) / 8, 256, 0, stream>>>(xs_in, resp_src, resp_dst,
                                                            agg_i, cnt_i, NE1);
        scatter_mean_acc<<<(NE1 + 7) / 8, 256, 0, stream>>>(xi_in, rev_src, rev_dst,
                                                            agg_s, cnt_s, NE1);
        // mean + bf16
        norm_to_bf16<<<gNI, EL, 0, stream>>>(agg_i, cnt_i, aggi_bf, NI * DD);
        norm_to_bf16<<<gNS, EL, 0, stream>>>(agg_s, cnt_s, aggs_bf, NS * DD);

        // SimpleConv: h = xs @ W_p^T  (reuse agg_s storage as h), then scatter
        gemm_plain<<<NS / 16, 256, 0, stream>>>(xs_bf, Wp_bf, agg_s);
        scatter_add<<<(NE2 + 7) / 8, 256, 0, stream>>>(agg_s, prec_src, prec_dst,
                                                       simple_acc, NE2);

        // item SAGE + ELU (reuse agg_i storage as elu(item))
        gemm_sage_item<<<NI / 16, 256, 0, stream>>>(aggi_bf, xi_bf, Wlri_bf, Wrri_bf,
                                                    bl_ri + bofs, agg_i);
        // batchnorm over items
        bn_stats<<<(NI + 127) / 128, 256, 0, stream>>>(agg_i, bnsum, bnsq, NI);
        bn_apply<<<gNI, EL, 0, stream>>>(agg_i, bnsum, bnsq, bn_g + bofs, bn_b + bofs,
                                         xi_out, NI);

        // student: 0.5*(SAGE + Simple)
        gemm_sage_stu<<<NS / 16, 256, 0, stream>>>(aggs_bf, xs_bf, Wlrs_bf, Wrrs_bf,
                                                   bl_rs + bofs, simple_acc,
                                                   b_p + bofs, xs_out);
    }
}
